// QuantumFeatureMappingLayer_39608188404038
// MI455X (gfx1250) — matrix operand order repaired and checked by reading, not compiled, not run on hardware
//
#include <hip/hip_runtime.h>
#include <math.h>

typedef __attribute__((ext_vector_type(8)))  float  v8f;
typedef __attribute__((ext_vector_type(4)))  __bf16 v4bf;
typedef __attribute__((ext_vector_type(8)))  __bf16 v8bf;
typedef __attribute__((ext_vector_type(16))) __bf16 v16bf;

#define N_ROWS 100000
#define CDIM   1024
#define ODIM   256
#define QDIM   2076
#define TOPK   8
#define NPAIR  28
#define EPSV   1e-5f
#define PI_F   3.14159265358979323846f
#define SBF    40     // LDS row stride in bf16: 80B -> 16B aligned, bank-step 20 (order 16 mod 64)

__constant__ int c_iu[NPAIR] = {0,0,0,0,0,0,0, 1,1,1,1,1,1, 2,2,2,2,2, 3,3,3,3, 4,4,4, 5,5, 6};
__constant__ int c_ju[NPAIR] = {1,2,3,4,5,6,7, 2,3,4,5,6,7, 3,4,5,6,7, 4,5,6,7, 5,6,7, 6,7, 7};

static __device__ __forceinline__ v8f wmma_bf16(v16bf a, v16bf b, v8f c) {
  // D(16x16 f32) = A(16x32 bf16) * B(32x16 bf16) + C
  return __builtin_amdgcn_wmma_f32_16x16x32_bf16(false, a, false, b, (short)0, c, false, false);
}

// assemble a v16bf fragment from two 16B-aligned 8-element runs in LDS
static __device__ __forceinline__ v16bf frag16(const __bf16* p0, const __bf16* p1) {
  v8bf a = *(const v8bf*)p0;
  v8bf b = *(const v8bf*)p1;
  return __builtin_shufflevector(a, b, 0,1,2,3,4,5,6,7,8,9,10,11,12,13,14,15);
}

// ---------------------------------------------------------------------------
// Kernel 0: zero the double-precision column-moment accumulators
// ---------------------------------------------------------------------------
__global__ void k_zero_aux(double* __restrict__ colsum, double* __restrict__ colsumsq) {
  int i = blockIdx.x * blockDim.x + threadIdx.x;
  if (i < CDIM) { colsum[i] = 0.0; colsumsq[i] = 0.0; }
}

// ---------------------------------------------------------------------------
// Kernel 1: z = x @ W_proj^T + b_proj
// bf16x3 split emulation on V_WMMA_F32_16X16X32_BF16 (fp32-grade accuracy,
// 8x MAC density vs the f32 K=4 WMMA). 128x128x32 tiles, 8 waves.
// ---------------------------------------------------------------------------
__global__ __launch_bounds__(256) void k_gemm1(
    const float* __restrict__ X, const float* __restrict__ W,
    const float* __restrict__ bias, float* __restrict__ Z)
{
  __shared__ __align__(16) __bf16 AsH[128][SBF];
  __shared__ __align__(16) __bf16 AsL[128][SBF];
  __shared__ __align__(16) __bf16 BsH[128][SBF];
  __shared__ __align__(16) __bf16 BsL[128][SBF];

  const int tid  = threadIdx.x;
  const int wave = tid >> 5;
  const int lane = tid & 31;
  const int m16  = lane & 15;
  const int hi   = lane >> 4;
  const int row0 = blockIdx.x * 128;
  const int col0 = blockIdx.y * 128;
  const int wm   = wave >> 1;   // 0..3 -> 32-row slice
  const int wn   = wave & 1;    // 0..1 -> 64-col slice

  v8f acc[2][4];
#pragma unroll
  for (int i = 0; i < 2; ++i)
#pragma unroll
    for (int j = 0; j < 4; ++j)
#pragma unroll
      for (int e = 0; e < 8; ++e) acc[i][j][e] = 0.0f;

  const int lr  = tid >> 3;        // 0..31
  const int lc4 = (tid & 7) * 4;   // 0,4,...,28

  for (int k0 = 0; k0 < CDIM; k0 += 32) {
#pragma unroll
    for (int it = 0; it < 4; ++it) {
      int r    = lr + it * 32;
      int grow = row0 + r;
      float4 va = make_float4(0.f, 0.f, 0.f, 0.f);
      if (grow < N_ROWS)
        va = *(const float4*)(X + (size_t)grow * CDIM + k0 + lc4);
      float4 vb = *(const float4*)(W + (size_t)(col0 + r) * CDIM + k0 + lc4);
      v4bf ah, al, bh, bl;
#pragma unroll
      for (int j = 0; j < 4; ++j) {
        float fa = (&va.x)[j];
        __bf16 h = (__bf16)fa;
        ah[j] = h; al[j] = (__bf16)(fa - (float)h);
        float fb = (&vb.x)[j];
        __bf16 g = (__bf16)fb;
        bh[j] = g; bl[j] = (__bf16)(fb - (float)g);
      }
      *(v4bf*)&AsH[r][lc4] = ah;  *(v4bf*)&AsL[r][lc4] = al;
      *(v4bf*)&BsH[r][lc4] = bh;  *(v4bf*)&BsL[r][lc4] = bl;
    }
    __syncthreads();

    // one K=32 WMMA step per chunk; A-frag: K {8hi..8hi+7} U {16+8hi..23+8hi}
    v16bf aH[2], aL[2];
#pragma unroll
    for (int mt = 0; mt < 2; ++mt) {
      const __bf16* pH = &AsH[wm * 32 + mt * 16 + m16][0];
      const __bf16* pL = &AsL[wm * 32 + mt * 16 + m16][0];
      aH[mt] = frag16(pH + 8 * hi, pH + 16 + 8 * hi);
      aL[mt] = frag16(pL + 8 * hi, pL + 16 + 8 * hi);
    }
#pragma unroll
    for (int nt = 0; nt < 4; ++nt) {
      const __bf16* pH = &BsH[wn * 64 + nt * 16 + m16][8 * hi];  // B-frag: K {8hi..8hi+7} U {16+8hi..23+8hi}
      const __bf16* pL = &BsL[wn * 64 + nt * 16 + m16][8 * hi];
      v16bf bH = frag16(pH, pH + 16);
      v16bf bL = frag16(pL, pL + 16);
#pragma unroll
      for (int mt = 0; mt < 2; ++mt) {
        acc[mt][nt] = wmma_bf16(aH[mt], bH, acc[mt][nt]);
        acc[mt][nt] = wmma_bf16(aL[mt], bH, acc[mt][nt]);
        acc[mt][nt] = wmma_bf16(aH[mt], bL, acc[mt][nt]);
      }
    }
    __syncthreads();
  }

#pragma unroll
  for (int mt = 0; mt < 2; ++mt) {
#pragma unroll
    for (int nt = 0; nt < 4; ++nt) {
      int col  = col0 + wn * 64 + nt * 16 + m16;
      float bv = bias[col];
#pragma unroll
      for (int e = 0; e < 8; ++e) {
        int grow = row0 + wm * 32 + mt * 16 + e + 8 * hi;   // C/D layout: M = vgpr + 8*hi
        if (grow < N_ROWS)
          Z[(size_t)grow * CDIM + col] = acc[mt][nt][e] + bv;
      }
    }
  }
}

// ---------------------------------------------------------------------------
// Kernel 2: row layernorm -> phi = pi*tanh(z) (in place); accumulate column
//           sum/sumsq of post-LN z in double (for top-k variance).
// ---------------------------------------------------------------------------
__global__ __launch_bounds__(256) void k_lnphi(
    float* __restrict__ Z, const float* __restrict__ g, const float* __restrict__ b,
    double* __restrict__ colsum, double* __restrict__ colsumsq)
{
  __shared__ double sAcc[2 * CDIM];
  const int tid = threadIdx.x;
  for (int i = tid; i < 2 * CDIM; i += 256) sAcc[i] = 0.0;
  __syncthreads();

  const int wave = tid >> 5;
  const int lane = tid & 31;
  const int rowBase = blockIdx.x * 128 + wave * 16;

  float4 g4[8], b4[8];
#pragma unroll
  for (int c = 0; c < 8; ++c) {
    int col = lane * 4 + c * 128;
    g4[c] = *(const float4*)(g + col);
    b4[c] = *(const float4*)(b + col);
  }

  float zs[32], zs2[32];
#pragma unroll
  for (int i = 0; i < 32; ++i) { zs[i] = 0.f; zs2[i] = 0.f; }

  for (int r = 0; r < 16; ++r) {
    int row = rowBase + r;
    if (row >= N_ROWS) break;          // uniform per wave
    float4 v[8];
    float s = 0.f, sq = 0.f;
#pragma unroll
    for (int c = 0; c < 8; ++c) {
      v[c] = *(const float4*)(Z + (size_t)row * CDIM + lane * 4 + c * 128);
#pragma unroll
      for (int j = 0; j < 4; ++j) {
        float x = (&v[c].x)[j];
        s += x; sq += x * x;
      }
    }
#pragma unroll
    for (int off = 16; off; off >>= 1) {
      s  += __shfl_xor(s,  off);
      sq += __shfl_xor(sq, off);
    }
    float mu   = s * (1.0f / CDIM);
    float var  = sq * (1.0f / CDIM) - mu * mu;
    float rstd = rsqrtf(var + EPSV);
#pragma unroll
    for (int c = 0; c < 8; ++c) {
      float4 ph;
#pragma unroll
      for (int j = 0; j < 4; ++j) {
        float zl = ((&v[c].x)[j] - mu) * rstd * (&g4[c].x)[j] + (&b4[c].x)[j];
        zs [c * 4 + j] += zl;
        zs2[c * 4 + j] += zl * zl;
        (&ph.x)[j] = PI_F * tanhf(zl);
      }
      *(float4*)(Z + (size_t)row * CDIM + lane * 4 + c * 128) = ph;
    }
  }

#pragma unroll
  for (int c = 0; c < 8; ++c) {
    int col = lane * 4 + c * 128;
#pragma unroll
    for (int j = 0; j < 4; ++j) {
      atomicAdd(&sAcc[col + j],        (double)zs [c * 4 + j]);   // ds_add_f64
      atomicAdd(&sAcc[CDIM + col + j], (double)zs2[c * 4 + j]);
    }
  }
  __syncthreads();
  for (int i = tid; i < 2 * CDIM; i += 256) {
    if (i < CDIM) atomicAdd(&colsum[i], sAcc[i]);                 // global_atomic_add_f64
    else          atomicAdd(&colsumsq[i - CDIM], sAcc[i]);
  }
}

// ---------------------------------------------------------------------------
// Kernel 3: unbiased column variance + serial top-8
// ---------------------------------------------------------------------------
__global__ void k_topk(const double* __restrict__ colsum, const double* __restrict__ colsumsq,
                       int* __restrict__ outIdx)
{
  __shared__ float var[CDIM];
  int tid = threadIdx.x;
  for (int i = tid; i < CDIM; i += 256) {
    double s = colsum[i];
    var[i] = (float)((colsumsq[i] - s * s / (double)N_ROWS) / (double)(N_ROWS - 1));
  }
  __syncthreads();
  if (tid == 0) {
    for (int t = 0; t < TOPK; ++t) {
      int best = 0; float bv = -1e30f;
      for (int i = 0; i < CDIM; ++i)
        if (var[i] > bv) { bv = var[i]; best = i; }   // first-max tie-break (matches top_k)
      outIdx[t] = best;
      var[best] = -1e30f;
    }
  }
}

// ---------------------------------------------------------------------------
// Kernel 4: fused q-generation + q @ W_out^T + b_out + row layernorm.
// Block = 16 rows x full 256 outputs; cos pass, sin pass, interaction pass,
// all on bf16x3-split V_WMMA_F32_16X16X32_BF16.
// ---------------------------------------------------------------------------
__global__ __launch_bounds__(256) void k_gemm2(
    const float* __restrict__ PHI, const float* __restrict__ Wout,
    const float* __restrict__ bout, const float* __restrict__ og,
    const float* __restrict__ ob, const float* __restrict__ alphaPtr,
    const int* __restrict__ selIdx, float* __restrict__ OutB)
{
  __shared__ __align__(16) __bf16 AsH[16][SBF];
  __shared__ __align__(16) __bf16 AsL[16][SBF];
  __shared__ __align__(16) __bf16 BsH[ODIM][SBF];   // reused as fp32 output tile in epilogue
  __shared__ __align__(16) __bf16 BsL[ODIM][SBF];
  __shared__ float selPhi[16][8];

  const int tid  = threadIdx.x;
  const int wave = tid >> 5;
  const int lane = tid & 31;
  const int m16  = lane & 15;
  const int hi   = lane >> 4;
  const int row0 = blockIdx.x * 16;               // N_ROWS = 6250*16, always in range
  const float alpha = alphaPtr[0];

  v8f acc[2];
#pragma unroll
  for (int t = 0; t < 2; ++t)
#pragma unroll
    for (int e = 0; e < 8; ++e) acc[t][e] = 0.0f;

  // cos pass (p=0) and sin pass (p=1)
  for (int p = 0; p < 2; ++p) {
    for (int k0 = 0; k0 < CDIM; k0 += 32) {
      {                                            // stage A: 16 x 32 q-values, split hi/lo
        int r  = tid >> 4;
        int c2 = (tid & 15) * 2;
        float2 ph = *(const float2*)(PHI + (size_t)(row0 + r) * CDIM + k0 + c2);
        float q0 = alpha * ((p == 0) ? cosf(ph.x) : sinf(ph.x));
        float q1 = alpha * ((p == 0) ? cosf(ph.y) : sinf(ph.y));
        __bf16 h0 = (__bf16)q0, h1 = (__bf16)q1;
        AsH[r][c2]     = h0;  AsL[r][c2]     = (__bf16)(q0 - (float)h0);
        AsH[r][c2 + 1] = h1;  AsL[r][c2 + 1] = (__bf16)(q1 - (float)h1);
      }
      {                                            // stage B: 256 x 32 of W_out, split hi/lo
        int r  = tid >> 3;
        int c4 = (tid & 7) * 4;
#pragma unroll
        for (int it = 0; it < 8; ++it) {
          int o = r + it * 32;
          float4 w4 = *(const float4*)(Wout + (size_t)o * QDIM + p * CDIM + k0 + c4);
          v4bf h4, l4;
#pragma unroll
          for (int j = 0; j < 4; ++j) {
            float f = (&w4.x)[j];
            __bf16 h = (__bf16)f;
            h4[j] = h; l4[j] = (__bf16)(f - (float)h);
          }
          *(v4bf*)&BsH[o][c4] = h4;
          *(v4bf*)&BsL[o][c4] = l4;
        }
      }
      __syncthreads();
      {
        const __bf16* paH = &AsH[m16][0];
        const __bf16* paL = &AsL[m16][0];
        v16bf aH = frag16(paH + 8 * hi, paH + 16 + 8 * hi);
        v16bf aL = frag16(paL + 8 * hi, paL + 16 + 8 * hi);
#pragma unroll
        for (int t = 0; t < 2; ++t) {
          const __bf16* pH = &BsH[wave * 32 + t * 16 + m16][8 * hi];
          const __bf16* pL = &BsL[wave * 32 + t * 16 + m16][8 * hi];
          v16bf bH = frag16(pH, pH + 16);
          v16bf bL = frag16(pL, pL + 16);
          acc[t] = wmma_bf16(aH, bH, acc[t]);
          acc[t] = wmma_bf16(aL, bH, acc[t]);
          acc[t] = wmma_bf16(aH, bL, acc[t]);
        }
      }
      __syncthreads();
    }
  }

  // interaction features: q_int = alpha * cos(phi_sel[iu] - phi_sel[ju]), 28 -> pad 32
  if (tid < 128) {
    int r = tid >> 3, s = tid & 7;
    selPhi[r][s] = PHI[(size_t)(row0 + r) * CDIM + selIdx[s]];
  }
  __syncthreads();
  for (int e = tid; e < 16 * 32; e += 256) {
    int r = e >> 5, pp = e & 31;
    float qv = 0.f;
    if (pp < NPAIR)
      qv = alpha * cosf(selPhi[r][c_iu[pp]] - selPhi[r][c_ju[pp]]);
    __bf16 h = (__bf16)qv;
    AsH[r][pp] = h;  AsL[r][pp] = (__bf16)(qv - (float)h);
  }
  {
    int o = tid;                                   // 0..255
    for (int pp = 0; pp < 32; ++pp) {
      float f = (pp < NPAIR) ? Wout[(size_t)o * QDIM + 2 * CDIM + pp] : 0.f;
      __bf16 h = (__bf16)f;
      BsH[o][pp] = h;  BsL[o][pp] = (__bf16)(f - (float)h);
    }
  }
  __syncthreads();
  {
    const __bf16* paH = &AsH[m16][0];
    const __bf16* paL = &AsL[m16][0];
    v16bf aH = frag16(paH + 8 * hi, paH + 16 + 8 * hi);
    v16bf aL = frag16(paL + 8 * hi, paL + 16 + 8 * hi);
#pragma unroll
    for (int t = 0; t < 2; ++t) {
      const __bf16* pH = &BsH[wave * 32 + t * 16 + m16][8 * hi];
      const __bf16* pL = &BsL[wave * 32 + t * 16 + m16][8 * hi];
      v16bf bH = frag16(pH, pH + 16);
      v16bf bL = frag16(pL, pL + 16);
      acc[t] = wmma_bf16(aH, bH, acc[t]);
      acc[t] = wmma_bf16(aL, bH, acc[t]);
      acc[t] = wmma_bf16(aH, bL, acc[t]);
    }
  }
  __syncthreads();

  // epilogue: bias, then fused row layernorm over 256 outputs via LDS tile
  float (*outT)[264] = (float(*)[264]) & BsH[0][0];   // 16*264*4 B = 16.9KB < 20.5KB, reuse
#pragma unroll
  for (int t = 0; t < 2; ++t) {
    int col  = wave * 32 + t * 16 + m16;
    float bv = bout[col];
#pragma unroll
    for (int e = 0; e < 8; ++e)
      outT[e + 8 * hi][col] = acc[t][e] + bv;
  }
  __syncthreads();

#pragma unroll
  for (int rr2 = 0; rr2 < 2; ++rr2) {
    int r = wave * 2 + rr2;
    float vals[8];
    float s = 0.f, sq = 0.f;
#pragma unroll
    for (int c = 0; c < 8; ++c) {
      vals[c] = outT[r][lane + 32 * c];
      s += vals[c]; sq += vals[c] * vals[c];
    }
#pragma unroll
    for (int off = 16; off; off >>= 1) { s += __shfl_xor(s, off); sq += __shfl_xor(sq, off); }
    float mu   = s * (1.0f / ODIM);
    float var  = sq * (1.0f / ODIM) - mu * mu;
    float rstd = rsqrtf(var + EPSV);
#pragma unroll
    for (int c = 0; c < 8; ++c) {
      int col = lane + 32 * c;
      OutB[(size_t)(row0 + r) * ODIM + col] = (vals[c] - mu) * rstd * og[col] + ob[col];
    }
  }
}

// ---------------------------------------------------------------------------
extern "C" void kernel_launch(void* const* d_in, const int* in_sizes, int n_in,
                              void* d_out, int out_size, void* d_ws, size_t ws_size,
                              hipStream_t stream) {
  (void)in_sizes; (void)n_in; (void)out_size; (void)ws_size;
  const float* x     = (const float*)d_in[0];
  const float* Wp    = (const float*)d_in[1];
  const float* bp    = (const float*)d_in[2];
  const float* lng   = (const float*)d_in[3];
  const float* lnb   = (const float*)d_in[4];
  const float* alpha = (const float*)d_in[5];
  const float* Wout  = (const float*)d_in[6];
  const float* bo    = (const float*)d_in[7];
  const float* og    = (const float*)d_in[8];
  const float* ob    = (const float*)d_in[9];
  float* out = (float*)d_out;

  char*   ws       = (char*)d_ws;
  float*  phi      = (float*)ws;                               // N*C fp32 (z, then phi in place)
  size_t  phiBytes = (size_t)N_ROWS * CDIM * sizeof(float);    // 409.6 MB
  double* colsum   = (double*)(ws + phiBytes);
  double* colsumsq = colsum + CDIM;
  int*    selIdx   = (int*)(colsumsq + CDIM);

  k_zero_aux<<<dim3(4), dim3(256), 0, stream>>>(colsum, colsumsq);

  dim3 g1((N_ROWS + 127) / 128, CDIM / 128);
  k_gemm1<<<g1, dim3(256), 0, stream>>>(x, Wp, bp, phi);

  k_lnphi<<<dim3((N_ROWS + 127) / 128), dim3(256), 0, stream>>>(phi, lng, lnb, colsum, colsumsq);

  k_topk<<<dim3(1), dim3(256), 0, stream>>>(colsum, colsumsq, selIdx);

  k_gemm2<<<dim3(N_ROWS / 16), dim3(256), 0, stream>>>(phi, Wout, bo, og, ob, alpha, selIdx, out);
}
